// DCNBlock_60129542144244
// MI455X (gfx1250) — matrix operand order repaired and checked by reading, not compiled, not run on hardware
//
#include <hip/hip_runtime.h>
#include <math.h>

typedef __attribute__((ext_vector_type(16))) _Float16 v16h;
typedef __attribute__((ext_vector_type(8)))  float    v8f;

#define BB 4
#define CC 64
#define OO 64
#define HH 192
#define WW 192
#define KP 9                  // kernel points
#define HW (HH*WW)            // 36864
#define CK (CC*KP)            // 576
#define KSTEPS (CK/32)        // 18
#define TILES_X (WW/16)       // 12
#define NTILES (BB*HH*TILES_X) // 9216

// packed-weight block sizes (bytes)
#define PB2_BYTES (18*4*32*16*2)   // 73728
#define PB1_BYTES (18*2*32*16*2)   // 36864

// workspace layout (bytes)
#define WS_PB2   0
#define WS_PB1   PB2_BYTES
#define WS_DY    (PB2_BYTES + PB1_BYTES)        // 110592
#define WS_DX    (WS_DY + 5308416)
#define WS_MK    (WS_DX + 5308416)

__device__ __forceinline__ v8f wmma16(v16h a, v16h b, v8f c) {
    // D = A(16x32 f16) * B(32x16 f16) + C(16x16 f32)
    return __builtin_amdgcn_wmma_f32_16x16x32_f16(false, a, false, b,
                                                  (short)0, c, false, false);
}

// ---------------------------------------------------------------------------
// TDM: DMA a contiguous block (nbytes, multiple of 8) from global to LDS.
// 1-D tile, data_size = 8B. Issued by one wave; tracked by TENSORcnt.
// D# layout per CDNA5 ISA ch.8 (group0 128b, group1 256b; groups 2/3 zero).
// This toolchain exposes the 6-arg builtin:
//   (u32x4 g0, i32x8 g1, i32x4 g2, i32x4 g3, i32x8 pad, i32 cpol)
// ---------------------------------------------------------------------------
__device__ __forceinline__ void tdm_load_block(const void* gsrc,
                                               unsigned lds_byte_off,
                                               unsigned nbytes) {
#if __has_builtin(__builtin_amdgcn_tensor_load_to_lds)
    typedef unsigned int u32x4 __attribute__((ext_vector_type(4)));
    typedef int          i32x8 __attribute__((ext_vector_type(8)));
    typedef int          i32x4 __attribute__((ext_vector_type(4)));
    unsigned n8 = nbytes >> 3;                       // elements of 8 bytes
    unsigned long long ga = (unsigned long long)gsrc;
    u32x4 g0;
    g0.x = 1u;                                       // count=1 (valid), user mode
    g0.y = lds_byte_off;                             // LDS dest byte address
    g0.z = (unsigned)(ga & 0xFFFFFFFFu);             // global_addr[31:0]
    g0.w = (unsigned)((ga >> 32) & 0x01FFFFFFu)      // global_addr[56:32]
         | (2u << 30);                               // type = 2 ("image")
    i32x8 g1;
    g1[0] = (int)(3u << 16);                         // wg_mask=0, data_size=8B
    g1[1] = (int)((n8 & 0xFFFFu) << 16);             // tensor_dim0[15:0]
    g1[2] = (int)((n8 >> 16) & 0xFFFFu) | (1 << 16); // dim0[31:16], tensor_dim1=1
    g1[3] = (int)((n8 & 0xFFFFu) << 16);             // tile_dim0 = n8 (<65536)
    g1[4] = 1;                                       // tile_dim1 = 1
    g1[5] = (int)n8;                                 // tensor_dim0_stride
    g1[6] = 0;
    g1[7] = 0;
    i32x4 z4 = {0, 0, 0, 0};
    i32x8 z8 = {0, 0, 0, 0, 0, 0, 0, 0};
    __builtin_amdgcn_tensor_load_to_lds(g0, g1, z4, z4, z8, 0);
#else
    (void)gsrc; (void)lds_byte_off; (void)nbytes;
#endif
}

__device__ __forceinline__ void tdm_wait0() {
#if __has_builtin(__builtin_amdgcn_s_wait_tensorcnt)
    __builtin_amdgcn_s_wait_tensorcnt(0);
#else
    asm volatile("s_wait_tensorcnt 0x0" ::: "memory");
#endif
}

// GEMM K ordering is kernel-point-major: kidx = kk*64 + c  (kk in 0..8, c in 0..63).
// Within K-step kt (32 wide): kk = kt>>1 (wave-uniform), c = (kt&1)*32 + K32.
// A-fragment K32 for element e:  K32 = e + (e>=8 ? 8:0) + (lane>=16 ? 8:0)
// B-fragment K32 for element e:  K32 = e + (e>=8 ? 8:0) + (lane>=16 ? 8:0)

// ---------------------------------------------------------------------------
// Kernel 0: pack weights (f32 -> f16) into per-lane WMMA B-fragment order,
// with the kk-major K permutation baked in.
// ---------------------------------------------------------------------------
__global__ __launch_bounds__(256)
void dcn_pack_kernel(const float* __restrict__ weight,    // [O][C][3][3] = [O][576]
                     const float* __restrict__ w_offset,  // [27][C][3][3] = [27][576]
                     _Float16* __restrict__ pB2,          // [18][4][32][16]
                     _Float16* __restrict__ pB1)          // [18][2][32][16]
{
    int idx = blockIdx.x * 256 + threadIdx.x;
    if (idx < 18 * 4 * 32 * 16) {
        int e    = idx & 15;
        int lane = (idx >> 4) & 31;
        int nt   = (idx >> 9) & 3;
        int kt   = idx >> 11;
        int kk   = kt >> 1;
        int c    = (kt & 1) * 32 + e + ((e >= 8) ? 8 : 0) + ((lane >= 16) ? 8 : 0);
        int N    = nt * 16 + (lane & 15);
        pB2[idx] = (_Float16)weight[N * CK + c * KP + kk];
    } else {
        int i    = idx - 18 * 4 * 32 * 16;
        int e    = i & 15;
        int lane = (i >> 4) & 31;
        int nt   = (i >> 9) & 1;
        int kt   = i >> 10;
        int kk   = kt >> 1;
        int c    = (kt & 1) * 32 + e + ((e >= 8) ? 8 : 0) + ((lane >= 16) ? 8 : 0);
        int N    = nt * 16 + (lane & 15);
        pB1[i]   = (N < 27) ? (_Float16)w_offset[N * CK + c * KP + kk]
                            : (_Float16)0.0f;
    }
}

// ---------------------------------------------------------------------------
// Kernel 1: offset conv (implicit GEMM, 16 px x 32(27) out x 576).
// Weights staged LDS-resident via TDM; one 16-px tile per wave, 8 waves/block.
// ---------------------------------------------------------------------------
__global__ __launch_bounds__(256)
void dcn_offsets_kernel(const float* __restrict__ x,
                        const _Float16* __restrict__ pB1,
                        const float* __restrict__ b_offset,
                        float* __restrict__ dyA,
                        float* __restrict__ dxA,
                        float* __restrict__ mkA)
{
    // [0, 36864): packed B1 fragments; [36864, 46080): tap records
    __shared__ __align__(16) char smem[PB1_BYTES + 8 * 144 * 8];
    v16h* sB   = (v16h*)smem;
    int2* taps = (int2*)(smem + PB1_BYTES);      // {clamped y*W+x, float 0/1 bits}

    const int lane = threadIdx.x & 31;
    const int wv   = threadIdx.x >> 5;
    const int tile = blockIdx.x * 8 + wv;
    const int b    = tile / (HH * TILES_X);
    const int rr   = tile % (HH * TILES_X);
    const int h    = rr / TILES_X;
    const int w0   = (rr % TILES_X) * 16;
    const int p    = lane & 15;                  // this lane's A row (pixel)
    const int kb   = (lane >= 16) ? 8 : 0;

#if __has_builtin(__builtin_amdgcn_tensor_load_to_lds)
    if (wv == 0 && lane == 0)
        tdm_load_block(pB1, 0u, PB1_BYTES);      // smem starts at LDS offset 0
#else
    for (int i = threadIdx.x; i < PB1_BYTES / 16; i += 256)
        ((uint4*)sB)[i] = ((const uint4*)pB1)[i];
#endif

    for (int r = lane; r < 144; r += 32) {
        int pp = r & 15, kk = r >> 4;
        int y  = h - 1 + kk / 3;
        int xx = w0 + pp - 1 + kk % 3;
        bool valid = (y >= 0 && y < HH && xx >= 0 && xx < WW);
        taps[wv * 144 + r] = make_int2(valid ? (y * WW + xx) : 0,
                                       __float_as_int(valid ? 1.0f : 0.0f));
    }
    tdm_wait0();
    __syncthreads();

    const float* xb = x + (size_t)b * CC * HW;

    v8f acc0 = {0.f,0.f,0.f,0.f,0.f,0.f,0.f,0.f};
    v8f acc1 = acc0;

    for (int kk2 = 0; kk2 < KP; ++kk2) {
        int2  t  = taps[wv * 144 + kk2 * 16 + p];
        float tw = __int_as_float(t.y);
        const float* xt = xb + t.x;
        #pragma unroll
        for (int half = 0; half < 2; ++half) {
            const int kt = kk2 * 2 + half;
            v16h bf0 = sB[(kt * 2 + 0) * 32 + lane];
            v16h bf1 = sB[(kt * 2 + 1) * 32 + lane];
            const int cbase = half * 32 + kb;
            v16h a;
            #pragma unroll
            for (int e = 0; e < 16; ++e) {
                int c = cbase + e + ((e >= 8) ? 8 : 0);
                a[e] = (_Float16)(xt[c * HW] * tw);
            }
            acc0 = wmma16(a, bf0, acc0);
            acc1 = wmma16(a, bf1, acc1);
        }
    }

    // Epilogue: D row M = v + 8*(lane/16) -> pixel; col N = lane%16 -> channel.
    const int chl  = lane & 15;
    const int prow = 8 * (lane >> 4);
    v8f accs[2] = {acc0, acc1};
    #pragma unroll
    for (int nt = 0; nt < 2; ++nt) {
        int ch = nt * 16 + chl;
        if (ch >= 27) continue;
        float bo = b_offset[ch];
        int wbase = w0 + prow;
        if (ch < 18) {
            int kidx = ch >> 1;
            float* dst = (ch & 1) ? dxA : dyA;
            float* pd  = dst + ((size_t)(b * KP + kidx) * HH + h) * WW + wbase;
            #pragma unroll
            for (int v = 0; v < 8; ++v) pd[v] = accs[nt][v] + bo;
        } else {
            int kidx = ch - 18;
            float* pm = mkA + ((size_t)(b * KP + kidx) * HH + h) * WW + wbase;
            #pragma unroll
            for (int v = 0; v < 8; ++v) {
                float z = accs[nt][v] + bo;
                pm[v] = 1.0f / (1.0f + __expf(-z));   // sigmoid
            }
        }
    }
}

// ---------------------------------------------------------------------------
// Kernel 2: main DCN GEMM: A = bilinear-sampled, mask-premultiplied values
// [16 px x 576], B = packed weights [576 x 64] (LDS-resident via TDM),
// D = [16 px x 64]. Dynamic LDS: 73728 (B) + 18432 (recI) + 18432 (recW).
// ---------------------------------------------------------------------------
__global__ __launch_bounds__(256)
void dcn_main_kernel(const float* __restrict__ x,
                     const _Float16* __restrict__ pB2,
                     const float* __restrict__ bias,
                     const float* __restrict__ dyA,
                     const float* __restrict__ dxA,
                     const float* __restrict__ mkA,
                     float* __restrict__ out)
{
    extern __shared__ __align__(16) char smem[];
    v16h*   sB   = (v16h*)smem;                          // 73728 B
    int4*   recI = (int4*)(smem + PB2_BYTES);            // 18432 B
    float4* recW = (float4*)(smem + PB2_BYTES + 18432);  // 18432 B

    const int lane = threadIdx.x & 31;
    const int wv   = threadIdx.x >> 5;
    const int tile = blockIdx.x * 8 + wv;
    const int b    = tile / (HH * TILES_X);
    const int rr   = tile % (HH * TILES_X);
    const int h    = rr / TILES_X;
    const int w0   = (rr % TILES_X) * 16;
    const int p    = lane & 15;
    const int kb   = (lane >= 16) ? 8 : 0;

#if __has_builtin(__builtin_amdgcn_tensor_load_to_lds)
    if (wv == 0 && lane == 0)
        tdm_load_block(pB2, 0u, PB2_BYTES);      // dynamic LDS starts at offset 0
#else
    for (int i = threadIdx.x; i < PB2_BYTES / 16; i += 256)
        ((uint4*)sB)[i] = ((const uint4*)pB2)[i];
#endif

    // Build 144 sampling records (16 px x 9 taps) cooperatively per wave.
    for (int r = lane; r < 144; r += 32) {
        int pp = r & 15, kk = r >> 4;
        int w  = w0 + pp;
        int gi = ((b * KP + kk) * HH + h) * WW + w;
        float dy = dyA[gi], dx = dxA[gi], mk = mkA[gi];
        float py = (float)(h - 1 + kk / 3) + dy;
        float px = (float)(w - 1 + kk % 3) + dx;
        float fy = floorf(py), fx = floorf(px);
        int y0 = (int)fy, x0 = (int)fx;
        int y1 = y0 + 1,  x1 = x0 + 1;
        float wy = py - fy, wx = px - fx;
        bool vy0 = (y0 >= 0) && (y0 < HH);
        bool vy1 = (y1 >= 0) && (y1 < HH);
        bool vx0 = (x0 >= 0) && (x0 < WW);
        bool vx1 = (x1 >= 0) && (x1 < WW);
        int y0c = min(max(y0, 0), HH - 1), y1c = min(max(y1, 0), HH - 1);
        int x0c = min(max(x0, 0), WW - 1), x1c = min(max(x1, 0), WW - 1);
        float w00 = (1.f - wy) * (1.f - wx) * mk; if (!(vy0 && vx0)) w00 = 0.f;
        float w01 = (1.f - wy) * wx        * mk; if (!(vy0 && vx1)) w01 = 0.f;
        float w10 = wy        * (1.f - wx) * mk; if (!(vy1 && vx0)) w10 = 0.f;
        float w11 = wy        * wx         * mk; if (!(vy1 && vx1)) w11 = 0.f;
        recI[wv * 144 + r] = make_int4(y0c * WW + x0c, y0c * WW + x1c,
                                       y1c * WW + x0c, y1c * WW + x1c);
        recW[wv * 144 + r] = make_float4(w00, w01, w10, w11);
    }
    tdm_wait0();
    __syncthreads();

    const float* xb = x + (size_t)b * CC * HW;

    v8f acc[4];
    #pragma unroll
    for (int nt = 0; nt < 4; ++nt) acc[nt] = (v8f){0.f,0.f,0.f,0.f,0.f,0.f,0.f,0.f};

    for (int kk2 = 0; kk2 < KP; ++kk2) {
        int4   ii  = recI[wv * 144 + kk2 * 16 + p];
        float4 wwv = recW[wv * 144 + kk2 * 16 + p];
        const float* x00 = xb + ii.x;
        const float* x01 = xb + ii.y;
        const float* x10 = xb + ii.z;
        const float* x11 = xb + ii.w;
        #pragma unroll
        for (int half = 0; half < 2; ++half) {
            const int kt = kk2 * 2 + half;
            v16h bf0 = sB[(kt * 4 + 0) * 32 + lane];
            v16h bf1 = sB[(kt * 4 + 1) * 32 + lane];
            v16h bf2 = sB[(kt * 4 + 2) * 32 + lane];
            v16h bf3 = sB[(kt * 4 + 3) * 32 + lane];

            const int cbase = half * 32 + kb;
            v16h a;
            #pragma unroll
            for (int e = 0; e < 16; ++e) {
                int coff = (cbase + e + ((e >= 8) ? 8 : 0)) * HW;
                float v = x00[coff] * wwv.x + x01[coff] * wwv.y +
                          x10[coff] * wwv.z + x11[coff] * wwv.w;
                a[e] = (_Float16)v;
            }
            acc[0] = wmma16(a, bf0, acc[0]);
            acc[1] = wmma16(a, bf1, acc[1]);
            acc[2] = wmma16(a, bf2, acc[2]);
            acc[3] = wmma16(a, bf3, acc[3]);
        }
    }

    // Epilogue: lane writes 8 consecutive pixels for its output channel.
    const int o_lo = lane & 15;
    const int prow = 8 * (lane >> 4);
    #pragma unroll
    for (int nt = 0; nt < 4; ++nt) {
        int o = nt * 16 + o_lo;
        float bo = bias[o];
        float* po = out + ((size_t)(b * OO + o)) * HW + (size_t)h * WW + w0 + prow;
        float4 lo = make_float4(acc[nt][0] + bo, acc[nt][1] + bo,
                                acc[nt][2] + bo, acc[nt][3] + bo);
        float4 hi = make_float4(acc[nt][4] + bo, acc[nt][5] + bo,
                                acc[nt][6] + bo, acc[nt][7] + bo);
        *(float4*)(po)     = lo;
        *(float4*)(po + 4) = hi;
    }
}

// ---------------------------------------------------------------------------
extern "C" void kernel_launch(void* const* d_in, const int* in_sizes, int n_in,
                              void* d_out, int out_size, void* d_ws, size_t ws_size,
                              hipStream_t stream) {
    const float* x        = (const float*)d_in[0];
    const float* w_offset = (const float*)d_in[1];
    const float* b_offset = (const float*)d_in[2];
    const float* weight   = (const float*)d_in[3];
    const float* bias     = (const float*)d_in[4];
    float* out = (float*)d_out;

    char* ws = (char*)d_ws;
    _Float16* pB2 = (_Float16*)(ws + WS_PB2);
    _Float16* pB1 = (_Float16*)(ws + WS_PB1);
    float*    dyA = (float*)(ws + WS_DY);
    float*    dxA = (float*)(ws + WS_DX);
    float*    mkA = (float*)(ws + WS_MK);

    // 1) pack weights into WMMA fragment order ((36864+18432)/256 = 216 blocks)
    dcn_pack_kernel<<<216, 256, 0, stream>>>(weight, w_offset, pB2, pB1);

    // 2) offset conv -> dy/dx/mask   (9216 tiles / 8 waves = 1152 blocks)
    dcn_offsets_kernel<<<NTILES / 8, 256, 0, stream>>>(x, pB1, b_offset,
                                                       dyA, dxA, mkA);

    // 3) main deformable-conv GEMM (dynamic LDS: weights + sampling records)
    dcn_main_kernel<<<NTILES / 8, 256, PB2_BYTES + 2 * 18432, stream>>>(
        x, pB2, bias, dyA, dxA, mkA, out);
}
